// Prediction_14044543058428
// MI455X (gfx1250) — matrix operand order repaired and checked by reading, not compiled, not run on hardware
//
#include <hip/hip_runtime.h>
#include <stdint.h>

typedef __attribute__((ext_vector_type(16))) __bf16 v16bf;
typedef __attribute__((ext_vector_type(8)))  __bf16 v8bf;
typedef __attribute__((ext_vector_type(8)))  float  v8f;

#define U_STEPS 128
#define NBATCH  512
#define CH      320
#define G4      1280                 // 4*CH
#define NC      (NBATCH*CH)          // 163840
#define UNC     ((size_t)U_STEPS*NC) // 20971520
#define WFRAG_L 819200               // per-layer swizzled weight elems: 20*20*4*32*16
#define CT_ELEMS 40960               // per-ct weight elems: 20(kb)*4(gate)*32(lane)*16

__device__ __forceinline__ __bf16 f2bf(float f) {
  union { float f; uint32_t u; } v; v.f = f;
  uint32_t r = v.u + 0x7FFFu + ((v.u >> 16) & 1u);   // round-to-nearest-even
  union { uint16_t s; __bf16 b; } o; o.s = (uint16_t)(r >> 16);
  return o.b;
}
__device__ __forceinline__ float fsigmoid(float x) {
  return 1.0f / (1.0f + __expf(-x));
}
__device__ __forceinline__ float ftanh(float x) {
  float e = __expf(-2.0f * fabsf(x));
  float t = (1.0f - e) / (1.0f + e);
  return copysignf(t, x);
}

// ---------------------------------------------------------------------------
// Swizzle w_hh (K-blocks 0..9) and w_ih (K-blocks 10..19) into the CDNA5
// 16-bit B-matrix (32x16) fragment lane layout, bf16.
// Flat layout: [layer][ct(20)][kb(20)][gate(4)][lane(32)][elem(16)]
// B lane layout used here (contiguous k, same as A): N = lane&15 ; K = (lane<16 ? 0 : 16) + elem
// ---------------------------------------------------------------------------
__global__ void prep_wfrag(const float* __restrict__ w_ih,
                           const float* __restrict__ w_hh,
                           __bf16* __restrict__ wfrag) {
  int idx = blockIdx.x * 256 + threadIdx.x;          // 1,638,400 total
  int i    = idx & 15;  int t = idx >> 4;
  int lane = t & 31;    t >>= 5;
  int gate = t & 3;     t >>= 2;
  int kb   = t % 20;    t /= 20;
  int ct   = t % 20;    t /= 20;
  int l    = t;                                      // 0..1
  int k_local = ((lane & 16) ? 16 : 0) + i;
  int n_local = lane & 15;
  int j = gate * CH + ct * 16 + n_local;             // gate column in [0,1280)
  int k = (kb % 10) * 32 + k_local;                  // source column in [0,320)
  const float* W = (kb < 10) ? w_hh : w_ih;          // kb<10: recurrent, else input
  wfrag[idx] = f2bf(W[((size_t)l * G4 + j) * CH + k]);
}

__global__ void prep_bias(const float* __restrict__ b_ih,
                          const float* __restrict__ b_hh,
                          float* __restrict__ bias) {
  int idx = blockIdx.x * 256 + threadIdx.x;          // 2*1280
  bias[idx] = b_ih[idx] + b_hh[idx];
}

// Embedding lookup -> bf16 input activations X0[U,N,C] (SOS=28 -> zeros)
__global__ void embed_kernel(const int* __restrict__ x,
                             const float* __restrict__ ew,
                             __bf16* __restrict__ X0) {
  size_t idx = (size_t)blockIdx.x * 256 + threadIdx.x; // UNC total
  int c = (int)(idx % CH);
  size_t tn = idx / CH;
  int n = (int)(tn % NBATCH);
  int t = (int)(tn / NBATCH);
  int lab = x[(size_t)t * NBATCH + n];
  float v = (lab == 28) ? 0.0f : ew[(size_t)lab * CH + c];
  X0[idx] = f2bf(v);
}

// Initialize recurrent state buffers from h0/c0 inputs
__global__ void init_state(const float* __restrict__ h0,
                           const float* __restrict__ c0,
                           __bf16* __restrict__ hping0,
                           __bf16* __restrict__ hping1,
                           float* __restrict__ c0_ws,
                           float* __restrict__ c1_ws,
                           float* __restrict__ h0f) {
  int idx = blockIdx.x * 256 + threadIdx.x;          // 2*NC
  int l = idx / NC;
  int r = idx % NC;
  __bf16 hb = f2bf(h0[idx]);
  if (l == 0) { hping0[r] = hb; c0_ws[r] = c0[idx]; h0f[r] = h0[idx]; }
  else        { hping1[r] = hb; c1_ws[r] = c0[idx]; }
}

// ---------------------------------------------------------------------------
// One LSTM time step, fused GEMM (K = 640 = [h(320) ; x_t(320)]) + cell.
// Block = 8 waves, all sharing one ct (16 hidden cols); each wave owns one
// nb (16 batch rows). The block's 80KB of swizzled B fragments are staged
// into LDS once via global_load_async_to_lds_b128, then the fully unrolled
// K loop issues 80 v_wmma_f32_16x16x32_bf16 per wave with A from global and
// B from LDS (ds_load_b128), so loads pipeline against the WMMAs.
// ---------------------------------------------------------------------------
__global__ void __launch_bounds__(256)
lstm_step(const __bf16* __restrict__ hbf_in,   // [N,C] recurrent input (bf16)
          const __bf16* __restrict__ xbf_in,   // [N,C] layer input (bf16)
          const __bf16* __restrict__ wfrag,    // this layer's swizzled weights
          const float*  __restrict__ bias,     // this layer's bias sum [1280]
          float*        __restrict__ c_ws,     // [N,C] cell state (in-place)
          float*        __restrict__ hf_out,   // [N,C] f32 h output
          __bf16*       __restrict__ hbf_out,  // [N,C] bf16 h (next step)
          __bf16*       __restrict__ xbf_out)  // [N,C] bf16 h (next layer) or null
{
  __shared__ __bf16 Bsh[CT_ELEMS];             // 80 KB

  const int tid   = threadIdx.x;
  const int lane  = tid & 31;
  const int wslot = tid >> 5;
  const int ct    = blockIdx.x % 20;
  const int nb    = (blockIdx.x / 20) * 8 + wslot;   // 0..31

  // ---- stage this ct's B fragments into LDS (async copy, 10x16B per thread)
  const __bf16* wsrc = wfrag + (size_t)ct * CT_ELEMS;
#pragma unroll
  for (int i = 0; i < 10; ++i) {
    const int chunk = i * 256 + tid;                 // 0..2559, 16 elems each
    uint64_t gaddr = (uint64_t)(uintptr_t)(wsrc + (size_t)chunk * 16);
    uint32_t laddr = (uint32_t)(uintptr_t)(Bsh + (size_t)chunk * 16);
    asm volatile("global_load_async_to_lds_b128 %0, %1, off"
                 :: "v"(laddr), "v"(gaddr) : "memory");
  }
  asm volatile("s_wait_asynccnt 0" ::: "memory");
  __syncthreads();

  v8f acc0 = {}, acc1 = {}, acc2 = {}, acc3 = {};

  // A fragment addressing (contiguous k order, same as the B swizzle):
  // lanes 0-15 hold K {b..b+15}; lanes 16-31 hold {b+16..b+31}
  const int rowA = nb * 16 + (lane & 15);
  const int koff = (lane >> 4) << 4;

#pragma unroll
  for (int half = 0; half < 2; ++half) {
    const __bf16* ap = (half ? xbf_in : hbf_in) + (size_t)rowA * CH + koff;
    const __bf16* bs = Bsh + half * 10 * 2048 + lane * 16;  // kb stride 2048, gate stride 512
#pragma unroll
    for (int k = 0; k < 10; ++k) {
      v8bf alo = *(const v8bf*)(ap);
      v8bf ahi = *(const v8bf*)(ap + 8);
      v16bf a = __builtin_shufflevector(alo, ahi,
                0,1,2,3,4,5,6,7,8,9,10,11,12,13,14,15);

      v16bf b0 = *(const v16bf*)(bs);
      v16bf b1 = *(const v16bf*)(bs + 512);
      v16bf b2 = *(const v16bf*)(bs + 1024);
      v16bf b3 = *(const v16bf*)(bs + 1536);

      acc0 = __builtin_amdgcn_wmma_f32_16x16x32_bf16(false, a, false, b0, (short)0, acc0, false, false);
      acc1 = __builtin_amdgcn_wmma_f32_16x16x32_bf16(false, a, false, b1, (short)0, acc1, false, false);
      acc2 = __builtin_amdgcn_wmma_f32_16x16x32_bf16(false, a, false, b2, (short)0, acc2, false, false);
      acc3 = __builtin_amdgcn_wmma_f32_16x16x32_bf16(false, a, false, b3, (short)0, acc3, false, false);

      ap += 32;
      bs += 2048;
    }
  }

  // Epilogue: C/D layout — VGPR r, lane l => row = nb*16 + r + 8*(l>=16), col = ct*16 + (l&15)
  const int col = ct * 16 + (lane & 15);
  const float bi = bias[col];
  const float bf = bias[CH + col];
  const float bg = bias[2 * CH + col];
  const float bo = bias[3 * CH + col];
  const int rbase = nb * 16 + ((lane >> 4) << 3);

#pragma unroll
  for (int r = 0; r < 8; ++r) {
    const size_t idx = (size_t)(rbase + r) * CH + col;
    float iv = fsigmoid(acc0[r] + bi);
    float fv = fsigmoid(acc1[r] + bf);
    float gv = ftanh   (acc2[r] + bg);
    float ov = fsigmoid(acc3[r] + bo);
    float cc = fv * c_ws[idx] + iv * gv;
    float hh = ov * ftanh(cc);
    c_ws[idx]   = cc;
    hf_out[idx] = hh;
    __bf16 hb = f2bf(hh);
    hbf_out[idx] = hb;
    if (xbf_out) xbf_out[idx] = hb;
  }
}

// Copy final hT/cT into output (hg[0]=layer0 hT, hg[1]=g[U-1], cg=c states)
__global__ void finalize(const float* __restrict__ g,
                         const float* __restrict__ h0f,
                         const float* __restrict__ c0_ws,
                         const float* __restrict__ c1_ws,
                         float* __restrict__ out_hg,
                         float* __restrict__ out_cg) {
  int idx = blockIdx.x * 256 + threadIdx.x;          // NC
  out_hg[idx]      = h0f[idx];
  out_hg[NC + idx] = g[(size_t)(U_STEPS - 1) * NC + idx];
  out_cg[idx]      = c0_ws[idx];
  out_cg[NC + idx] = c1_ws[idx];
}

extern "C" void kernel_launch(void* const* d_in, const int* in_sizes, int n_in,
                              void* d_out, int out_size, void* d_ws, size_t ws_size,
                              hipStream_t stream) {
  const int*   x    = (const int*)d_in[0];
  const float* h0   = (const float*)d_in[1];
  const float* c0   = (const float*)d_in[2];
  const float* ew   = (const float*)d_in[3];
  const float* w_ih = (const float*)d_in[4];
  const float* w_hh = (const float*)d_in[5];
  const float* b_ih = (const float*)d_in[6];
  const float* b_hh = (const float*)d_in[7];

  float* g_out = (float*)d_out;                 // [U,N,C]
  float* hg    = g_out + UNC;                   // [2,N,C]
  float* cg    = hg + 2 * (size_t)NC;           // [2,N,C]

  // Carve workspace (~49 MB)
  uint8_t* wp = (uint8_t*)d_ws;
  auto carve = [&](size_t bytes) -> void* {
    void* p = (void*)wp;
    wp += (bytes + 255) & ~(size_t)255;
    return p;
  };
  __bf16* wfrag = (__bf16*)carve(2ull * WFRAG_L * sizeof(__bf16));
  float*  bias  = (float*) carve(2ull * G4 * sizeof(float));
  __bf16* X0    = (__bf16*)carve(UNC * sizeof(__bf16));
  __bf16* x1    = (__bf16*)carve((size_t)NC * sizeof(__bf16));
  __bf16* hp[2][2];
  for (int l = 0; l < 2; ++l)
    for (int p = 0; p < 2; ++p)
      hp[l][p] = (__bf16*)carve((size_t)NC * sizeof(__bf16));
  float* c_ws0 = (float*)carve((size_t)NC * sizeof(float));
  float* c_ws1 = (float*)carve((size_t)NC * sizeof(float));
  float* h0f   = (float*)carve((size_t)NC * sizeof(float));

  prep_wfrag <<<6400, 256, 0, stream>>>(w_ih, w_hh, wfrag);
  prep_bias  <<<10,   256, 0, stream>>>(b_ih, b_hh, bias);
  embed_kernel<<<81920,256, 0, stream>>>(x, ew, X0);
  init_state <<<1280, 256, 0, stream>>>(h0, c0, hp[0][0], hp[1][0],
                                        c_ws0, c_ws1, h0f);

  int p0 = 0, p1 = 0;
  for (int t = 0; t < U_STEPS; ++t) {
    // layer 0: input = embeddings; f32 h -> h0f; bf16 h -> ping-pong + x1
    lstm_step<<<80, 256, 0, stream>>>(hp[0][p0], X0 + (size_t)t * NC,
                                      wfrag, bias,
                                      c_ws0, h0f, hp[0][p0 ^ 1], x1);
    p0 ^= 1;
    // layer 1: input = x1; f32 h -> final output g[t]; bf16 h -> ping-pong
    lstm_step<<<80, 256, 0, stream>>>(hp[1][p1], x1,
                                      wfrag + WFRAG_L, bias + G4,
                                      c_ws1, g_out + (size_t)t * NC,
                                      hp[1][p1 ^ 1], (__bf16*)nullptr);
    p1 ^= 1;
  }

  finalize<<<640, 256, 0, stream>>>(g_out, h0f, c_ws0, c_ws1, hg, cg);
}
